// SlidingWindowAttention_86517821216240
// MI455X (gfx1250) — matrix operand order repaired and checked by reading, not compiled, not run on hardware
//
#include <hip/hip_runtime.h>
#include <hip/hip_bf16.h>

// ---------------------------------------------------------------------------
// Sliding-window attention (D=1024, H=16, hd=64, WINDOW=128, B=2, S=2048)
// MI455X / gfx1250: wave32, WMMA 16x16x32 bf16, async global->LDS staging.
//
// Pipeline:
//   1) cast x, Wq, Wk, Wv, Wo  fp32 -> bf16
//   2) gemm_qkv: Q,K -> [B,H,S,64] bf16 (Q pre-scaled by 1/8),
//                V   -> [B,H,64,S+pad] bf16 (transposed, for P*V B-matrix)
//   3) attn: 4-wave WG per 64 queries; K/V band async-staged to LDS;
//            scores+softmax in registers; P*V via WMMA; K-LDS reused for P
//   4) gemm_out: Y @ Wo^T -> fp32 d_out
// ---------------------------------------------------------------------------

typedef __bf16 bf16_t;
typedef __attribute__((ext_vector_type(16))) __bf16 v16bf;
typedef __attribute__((ext_vector_type(8)))  __bf16 v8bf;
typedef __attribute__((ext_vector_type(4)))  __bf16 v4bf;
typedef __attribute__((ext_vector_type(8)))  float  v8f;

#define D_MODEL   1024
#define NHEADS    16
#define HDIM      64
#define SEQ       2048
#define BATCH     2
#define NROWS     (BATCH * SEQ)     // 4096
#define WND       128
#define VT_STRIDE 2080              // SEQ + pad
#define BAND      192               // shared key band per 64-query block
#define NKT_MAX   12                // BAND / 16

// ---------------- workspace layout (bytes) ----------------
#define WS_XB   ((size_t)0)
#define WS_WQB  (WS_XB  + (size_t)NROWS * D_MODEL * 2)
#define WS_WKB  (WS_WQB + (size_t)D_MODEL * D_MODEL * 2)
#define WS_WVB  (WS_WKB + (size_t)D_MODEL * D_MODEL * 2)
#define WS_WOB  (WS_WVB + (size_t)D_MODEL * D_MODEL * 2)
#define WS_QB   (WS_WOB + (size_t)D_MODEL * D_MODEL * 2)
#define WS_KB   (WS_QB  + (size_t)NROWS * D_MODEL * 2)
#define WS_VT   (WS_KB  + (size_t)NROWS * D_MODEL * 2)
#define WS_YB   (WS_VT  + (size_t)BATCH * NHEADS * HDIM * VT_STRIDE * 2)
#define WS_END  (WS_YB  + (size_t)NROWS * D_MODEL * 2)   // ~50.5 MB

// ---------------- helpers ----------------
static __device__ __forceinline__ v8bf ld8(const bf16_t* p) {
    return *(const v8bf*)p;            // global_load_b128 / ds_load_b128
}
static __device__ __forceinline__ v16bf cat16(v8bf lo, v8bf hi) {
    v16bf r;
#pragma unroll
    for (int i = 0; i < 8; ++i) { r[i] = lo[i]; r[i + 8] = hi[i]; }
    return r;
}
static __device__ __forceinline__ v8f wmma_bf16(v16bf a, v16bf b, v8f c) {
    return __builtin_amdgcn_wmma_f32_16x16x32_bf16(
        false, a, false, b, (short)0, c, false, false);
}
// CDNA5 async global->LDS copy (ASYNCcnt-tracked); per-lane 16B.
// vdst gives the LDS byte address; generic shared ptr low 32 bits == LDS offset.
static __device__ __forceinline__ void async_ld_b128(unsigned ldsoff, const void* g) {
    asm volatile("global_load_async_to_lds_b128 %0, %1, off"
                 :: "v"(ldsoff), "v"(g) : "memory");
}
static __device__ __forceinline__ void wait_async0() {
    asm volatile("s_wait_asynccnt 0x0" ::: "memory");
}
static __device__ __forceinline__ unsigned lds_off(const void* p) {
    return (unsigned)(unsigned long long)p;   // low 32 bits of LDS aperture addr
}

// ---------------- 1) fp32 -> bf16 cast ----------------
__global__ void cast_f32_bf16(const float* __restrict__ src,
                              bf16_t* __restrict__ dst, int n4) {
    int i = blockIdx.x * 256 + threadIdx.x;
    if (i >= n4) return;
    float4 v = reinterpret_cast<const float4*>(src)[i];
    v4bf o;
    o[0] = (bf16_t)v.x; o[1] = (bf16_t)v.y; o[2] = (bf16_t)v.z; o[3] = (bf16_t)v.w;
    reinterpret_cast<v4bf*>(dst)[i] = o;
}

// ---------------- 2) QKV projection GEMM ----------------
// C[m,n] = sum_k X[m,k] * W[n,k]. 8 waves (4M x 2N), wave tile 32x64.
// blockIdx.z: 0=Q (scale 1/8), 1=K, 2=V(transposed).
__global__ __launch_bounds__(256) void gemm_qkv(
    const bf16_t* __restrict__ Xb,
    const bf16_t* __restrict__ Wqb, const bf16_t* __restrict__ Wkb,
    const bf16_t* __restrict__ Wvb,
    bf16_t* __restrict__ Qb, bf16_t* __restrict__ Kb, bf16_t* __restrict__ Vt) {
    const int lane = threadIdx.x & 31;
    const int wave = threadIdx.x >> 5;
    const int l15 = lane & 15, lh = lane >> 4;
    const int mode = blockIdx.z;
    const bf16_t* W = (mode == 0) ? Wqb : (mode == 1) ? Wkb : Wvb;
    const int m_base = blockIdx.x * 128 + (wave >> 1) * 32;
    const int n_base = blockIdx.y * 128 + (wave & 1) * 64;

    v8f acc[2][4];
#pragma unroll
    for (int t = 0; t < 2; ++t)
#pragma unroll
        for (int j = 0; j < 4; ++j) acc[t][j] = v8f{};

    const int ca = lh * 8, cb = lh * 8;
    const bf16_t* Arow0 = Xb + (size_t)(m_base + l15) * D_MODEL;
    const bf16_t* Arow1 = Arow0 + (size_t)16 * D_MODEL;
    const bf16_t* Wrow[4];
#pragma unroll
    for (int j = 0; j < 4; ++j)
        Wrow[j] = W + (size_t)(n_base + j * 16 + l15) * D_MODEL;

    for (int k0 = 0; k0 < D_MODEL; k0 += 32) {
        __builtin_prefetch(Arow0 + k0 + 128, 0, 1);
        __builtin_prefetch(Wrow[0] + k0 + 128, 0, 1);
        v16bf a0 = cat16(ld8(Arow0 + k0 + ca), ld8(Arow0 + k0 + 16 + ca));
        v16bf a1 = cat16(ld8(Arow1 + k0 + ca), ld8(Arow1 + k0 + 16 + ca));
        v16bf b[4];
#pragma unroll
        for (int j = 0; j < 4; ++j)
            b[j] = cat16(ld8(Wrow[j] + k0 + cb), ld8(Wrow[j] + k0 + cb + 16));
#pragma unroll
        for (int j = 0; j < 4; ++j) {
            acc[0][j] = wmma_bf16(a0, b[j], acc[0][j]);
            acc[1][j] = wmma_bf16(a1, b[j], acc[1][j]);
        }
    }

    // ---- epilogue: block stays inside one batch (2048 % 128 == 0) ----
    const int bb = m_base >> 11;
    const int s0 = (m_base & (SEQ - 1)) + lh * 8;   // + t*16 + r
    if (mode == 2) {
        // Vt[(bb*H+head)*64 + d][s]; r-consecutive in s -> one b128 store/tile
#pragma unroll
        for (int t = 0; t < 2; ++t)
#pragma unroll
            for (int j = 0; j < 4; ++j) {
                int n = n_base + j * 16 + l15;
                size_t row = (size_t)((bb * NHEADS + (n >> 6)) * HDIM + (n & 63));
                v8bf o;
#pragma unroll
                for (int r = 0; r < 8; ++r) o[r] = (bf16_t)acc[t][j][r];
                *(v8bf*)&Vt[row * VT_STRIDE + s0 + t * 16] = o;
            }
    } else {
        bf16_t* dst = (mode == 0) ? Qb : Kb;
        const float scale = (mode == 0) ? 0.125f : 1.0f;
#pragma unroll
        for (int t = 0; t < 2; ++t)
#pragma unroll
            for (int j = 0; j < 4; ++j) {
                int n = n_base + j * 16 + l15;
                size_t base =
                    ((size_t)((bb * NHEADS + (n >> 6)) * SEQ + s0 + t * 16)) * HDIM +
                    (n & 63);
#pragma unroll
                for (int r = 0; r < 8; ++r)
                    dst[base + (size_t)r * HDIM] = (bf16_t)(acc[t][j][r] * scale);
            }
    }
}

// ---------------- 3) sliding-window attention ----------------
// WG = 4 waves = 64 queries; shared 192-key band async-staged into LDS.
__global__ __launch_bounds__(128) void attn(
    const bf16_t* __restrict__ Qb, const bf16_t* __restrict__ Kb,
    const bf16_t* __restrict__ Vt, bf16_t* __restrict__ Yb) {
    __shared__ __align__(16) bf16_t s_kp[BAND * HDIM];   // 24 KB: K band, then P
    __shared__ __align__(16) bf16_t s_v[HDIM * BAND];    // 24 KB: V band

    const int tid  = threadIdx.x;
    const int lane = tid & 31;
    const int wave = tid >> 5;
    const int l15 = lane & 15, lh = lane >> 4;
    const int q0 = blockIdx.x * 64;
    const int h  = blockIdx.y;
    const int bb = blockIdx.z;
    const size_t qk_base = (size_t)(bb * NHEADS + h) * SEQ;
    const size_t vt_base = (size_t)(bb * NHEADS + h) * HDIM;

    const int kt_lo  = (q0 >= WND) ? (q0 >> 4) - 8 : 0;
    const int nkt    = (q0 >> 4) + 4 - kt_lo;            // 4..12, WG-uniform
    const int kbase0 = kt_lo * 16;

    // ---- async stage K band (contiguous in Kb) and V band into LDS ----
    {
        const char* ksrc = (const char*)(Kb + (qk_base + kbase0) * HDIM);
        unsigned kdst = lds_off(&s_kp[0]);
        const int kunits = nkt * 128;                    // 16B units
        for (int u = tid; u < kunits; u += 128)
            async_ld_b128(kdst + u * 16, ksrc + (size_t)u * 16);

        const char* vsrc = (const char*)(Vt + vt_base * VT_STRIDE + kbase0);
        unsigned vdst = lds_off(&s_v[0]);
        for (int u = tid; u < HDIM * (BAND / 8); u += 128) {  // 64*24 units
            int row = u / (BAND / 8);
            int c   = u % (BAND / 8);
            async_ld_b128(vdst + u * 16,
                          vsrc + (size_t)row * VT_STRIDE * 2 + c * 16);
        }
    }

    // Q tile for this wave (overlaps the async DMA)
    const int qw = q0 + wave * 16;
    const bf16_t* qrow = Qb + (qk_base + qw + l15) * HDIM;
    v16bf aq0 = cat16(ld8(qrow + lh * 8),      ld8(qrow + 16 + lh * 8));
    v16bf aq1 = cat16(ld8(qrow + 32 + lh * 8), ld8(qrow + 48 + lh * 8));

    wait_async0();
    __syncthreads();

    // ---- scores: Q K^T, mask folded in; tiles kept in registers ----
    v8f cs[NKT_MAX];
#pragma unroll
    for (int kt = 0; kt < NKT_MAX; ++kt) {
        if (kt >= nkt) continue;                  // uniform guard, EXEC stays full
        const bf16_t* krow = &s_kp[(size_t)(kt * 16 + l15) * HDIM + lh * 8];
        v16bf bk0 = cat16(ld8(krow),      ld8(krow + 16));
        v16bf bk1 = cat16(ld8(krow + 32), ld8(krow + 48));
        v8f t = v8f{};
        t = wmma_bf16(aq0, bk0, t);
        t = wmma_bf16(aq1, bk1, t);
#pragma unroll
        for (int r = 0; r < 8; ++r) {
            int q   = qw + lh * 8 + r;
            int key = kbase0 + kt * 16 + l15;
            int rel = q - key;
            t[r] = (rel >= 0 && rel < WND) ? t[r] : -1e30f;
        }
        cs[kt] = t;
    }

    // ---- softmax in registers (row split across the 16 l15 lanes) ----
    v8f mx;
#pragma unroll
    for (int r = 0; r < 8; ++r) mx[r] = -1e30f;
#pragma unroll
    for (int kt = 0; kt < NKT_MAX; ++kt) {
        if (kt >= nkt) continue;
#pragma unroll
        for (int r = 0; r < 8; ++r) mx[r] = fmaxf(mx[r], cs[kt][r]);
    }
#pragma unroll
    for (int s = 1; s < 16; s <<= 1)
#pragma unroll
        for (int r = 0; r < 8; ++r) mx[r] = fmaxf(mx[r], __shfl_xor(mx[r], s, 32));

    v8f sm;
#pragma unroll
    for (int r = 0; r < 8; ++r) sm[r] = 0.f;
#pragma unroll
    for (int kt = 0; kt < NKT_MAX; ++kt) {
        if (kt >= nkt) continue;
#pragma unroll
        for (int r = 0; r < 8; ++r) {
            float e = __expf(cs[kt][r] - mx[r]);
            cs[kt][r] = e;
            sm[r] += e;
        }
    }
#pragma unroll
    for (int s = 1; s < 16; s <<= 1)
#pragma unroll
        for (int r = 0; r < 8; ++r) sm[r] += __shfl_xor(sm[r], s, 32);
#pragma unroll
    for (int r = 0; r < 8; ++r) sm[r] = 1.0f / sm[r];   // diagonal allowed => >0

    __syncthreads();          // everyone done reading K band; reuse LDS for P

    bf16_t* pw = &s_kp[(size_t)wave * 16 * BAND];
#pragma unroll
    for (int kt = 0; kt < NKT_MAX; ++kt)
#pragma unroll
        for (int r = 0; r < 8; ++r) {
            float p = (kt < nkt) ? cs[kt][r] * sm[r] : 0.f;
            pw[(size_t)(lh * 8 + r) * BAND + kt * 16 + l15] = (bf16_t)p;
        }

    // P A-matrices (6 K=32 chunks); same-wave LDS ops are in-order
    v16bf ap[6];
#pragma unroll
    for (int c = 0; c < 6; ++c) {
        const bf16_t* pr = &pw[(size_t)l15 * BAND + c * 32 + lh * 8];
        ap[c] = cat16(*(const v8bf*)pr, *(const v8bf*)(pr + 16));
    }

#pragma unroll
    for (int j = 0; j < 4; ++j) {
        v8f cy = v8f{};
#pragma unroll
        for (int c = 0; c < 6; ++c) {
            const bf16_t* vr = &s_v[(size_t)(j * 16 + l15) * BAND + c * 32 + lh * 8];
            v16bf bv = cat16(ld8(vr), ld8(vr + 16));
            cy = wmma_bf16(ap[c], bv, cy);
        }
#pragma unroll
        for (int r = 0; r < 8; ++r) {
            int srow = qw + lh * 8 + r;
            int col  = h * HDIM + j * 16 + l15;
            Yb[((size_t)(bb * SEQ + srow)) * D_MODEL + col] = (bf16_t)cy[r];
        }
    }
}

// ---------------- 4) output projection ----------------
__global__ __launch_bounds__(256) void gemm_out(
    const bf16_t* __restrict__ Yb, const bf16_t* __restrict__ Wob,
    float* __restrict__ out) {
    const int lane = threadIdx.x & 31;
    const int wave = threadIdx.x >> 5;
    const int l15 = lane & 15, lh = lane >> 4;
    const int m_base = blockIdx.x * 128 + (wave >> 1) * 32;
    const int n_base = blockIdx.y * 128 + (wave & 1) * 64;

    v8f acc[2][4];
#pragma unroll
    for (int t = 0; t < 2; ++t)
#pragma unroll
        for (int j = 0; j < 4; ++j) acc[t][j] = v8f{};

    const int ca = lh * 8, cb = lh * 8;
    const bf16_t* Arow0 = Yb + (size_t)(m_base + l15) * D_MODEL;
    const bf16_t* Arow1 = Arow0 + (size_t)16 * D_MODEL;
    const bf16_t* Wrow[4];
#pragma unroll
    for (int j = 0; j < 4; ++j)
        Wrow[j] = Wob + (size_t)(n_base + j * 16 + l15) * D_MODEL;

    for (int k0 = 0; k0 < D_MODEL; k0 += 32) {
        __builtin_prefetch(Arow0 + k0 + 128, 0, 1);
        v16bf a0 = cat16(ld8(Arow0 + k0 + ca), ld8(Arow0 + k0 + 16 + ca));
        v16bf a1 = cat16(ld8(Arow1 + k0 + ca), ld8(Arow1 + k0 + 16 + ca));
        v16bf b[4];
#pragma unroll
        for (int j = 0; j < 4; ++j)
            b[j] = cat16(ld8(Wrow[j] + k0 + cb), ld8(Wrow[j] + k0 + cb + 16));
#pragma unroll
        for (int j = 0; j < 4; ++j) {
            acc[0][j] = wmma_bf16(a0, b[j], acc[0][j]);
            acc[1][j] = wmma_bf16(a1, b[j], acc[1][j]);
        }
    }
#pragma unroll
    for (int t = 0; t < 2; ++t)
#pragma unroll
        for (int j = 0; j < 4; ++j) {
            size_t base = (size_t)(m_base + t * 16 + lh * 8) * D_MODEL +
                          n_base + j * 16 + l15;
#pragma unroll
            for (int r = 0; r < 8; ++r)
                out[base + (size_t)r * D_MODEL] = acc[t][j][r];
        }
}

// ---------------- launch ----------------
extern "C" void kernel_launch(void* const* d_in, const int* in_sizes, int n_in,
                              void* d_out, int out_size, void* d_ws, size_t ws_size,
                              hipStream_t stream) {
    const float* x  = (const float*)d_in[0];
    const float* Wq = (const float*)d_in[1];
    const float* Wk = (const float*)d_in[2];
    const float* Wv = (const float*)d_in[3];
    const float* Wo = (const float*)d_in[4];
    float* out = (float*)d_out;
    char* ws = (char*)d_ws;
    bf16_t* xb  = (bf16_t*)(ws + WS_XB);
    bf16_t* wqb = (bf16_t*)(ws + WS_WQB);
    bf16_t* wkb = (bf16_t*)(ws + WS_WKB);
    bf16_t* wvb = (bf16_t*)(ws + WS_WVB);
    bf16_t* wob = (bf16_t*)(ws + WS_WOB);
    bf16_t* Qb  = (bf16_t*)(ws + WS_QB);
    bf16_t* Kb  = (bf16_t*)(ws + WS_KB);
    bf16_t* Vt  = (bf16_t*)(ws + WS_VT);
    bf16_t* Yb  = (bf16_t*)(ws + WS_YB);

    const int nx4 = NROWS * D_MODEL / 4;
    const int nw4 = D_MODEL * D_MODEL / 4;
    cast_f32_bf16<<<(nx4 + 255) / 256, 256, 0, stream>>>(x,  xb,  nx4);
    cast_f32_bf16<<<(nw4 + 255) / 256, 256, 0, stream>>>(Wq, wqb, nw4);
    cast_f32_bf16<<<(nw4 + 255) / 256, 256, 0, stream>>>(Wk, wkb, nw4);
    cast_f32_bf16<<<(nw4 + 255) / 256, 256, 0, stream>>>(Wv, wvb, nw4);
    cast_f32_bf16<<<(nw4 + 255) / 256, 256, 0, stream>>>(Wo, wob, nw4);

    gemm_qkv<<<dim3(NROWS / 128, D_MODEL / 128, 3), 256, 0, stream>>>(
        xb, wqb, wkb, wvb, Qb, Kb, Vt);

    attn<<<dim3(SEQ / 64, NHEADS, BATCH), 128, 0, stream>>>(Qb, Kb, Vt, Yb);

    gemm_out<<<dim3(NROWS / 128, D_MODEL / 128, 1), 256, 0, stream>>>(Yb, wob, out);
}
